// IGAB_8022998909124
// MI455X (gfx1250) — matrix operand order repaired and checked by reading, not compiled, not run on hardware
//
#include <hip/hip_runtime.h>
#include <hip/hip_bf16.h>

typedef __attribute__((ext_vector_type(16))) _Float16 v16h;
typedef __attribute__((ext_vector_type(8)))  _Float16 v8h;
typedef __attribute__((ext_vector_type(8)))  float    v8f;

#define HW    16384
#define DIMC  256
#define M2    32768   // 2 * HW
#define NBLK  2

__device__ __forceinline__ float gelu_exact(float x) {
    return 0.5f * x * (1.0f + erff(x * 0.70710678118654752440f));
}

// CDNA5 async global->LDS copy (ASYNCcnt-tracked), per-lane 16B.
__device__ __forceinline__ void async_ld_b128(unsigned lds_off, const _Float16* g) {
    asm volatile("global_load_async_to_lds_b128 %0, %1, off"
                 :: "v"(lds_off), "v"(g) : "memory");
}
__device__ __forceinline__ void wait_async0() {
    asm volatile("s_wait_asynccnt 0x0" ::: "memory");
}

// ---------------- layout transforms ----------------
__global__ __launch_bounds__(256) void k_nchw_to_nhwc(const float* __restrict__ in,
                                                      float* __restrict__ out) {
    int gid = blockIdx.x * 256 + threadIdx.x;          // (b*HW+p)*256 + c
    int c = gid & 255;
    int p = (gid >> 8) & (HW - 1);
    int b = gid >> 22;
    out[gid] = in[((size_t)(b * DIMC + c)) * HW + p];
}

__global__ __launch_bounds__(256) void k_nhwc_to_nchw(const float* __restrict__ in,
                                                      float* __restrict__ out) {
    int gid = blockIdx.x * 256 + threadIdx.x;          // (b*256+c)*HW + p
    int p = gid & (HW - 1);
    int c = (gid >> 14) & 255;
    int b = gid >> 22;
    out[gid] = in[((size_t)(b * HW + p)) * DIMC + c];
}

__global__ __launch_bounds__(256) void k_f32_to_f16(const float* __restrict__ in,
                                                    _Float16* __restrict__ out, int n) {
    int i = blockIdx.x * 256 + threadIdx.x;
    if (i < n) out[i] = (_Float16)in[i];
}

// W [K,N] f32 row-major  ->  Wt [N,K] f16 row-major
__global__ __launch_bounds__(256) void k_stage_wt(const float* __restrict__ W,
                                                  _Float16* __restrict__ Wt, int K, int N) {
    int i = blockIdx.x * 256 + threadIdx.x;
    if (i >= K * N) return;
    int n = i / K;
    int k = i - n * K;
    Wt[i] = (_Float16)W[k * N + n];
}

// ---------------- WMMA GEMM: C[M,N] = A[M,K] * Bt[N,K]^T ----------------
// Workgroup tile 128x128, 8 waves each 32x64 (2x4 v_wmma_f32_16x16x32_f16).
// Double-buffered async global->LDS staging (ASYNCcnt + one barrier per K-step).
// EPI: 0 = f32 store, 1 = f32 + bias + add0 + add1, 2 = f16 + gelu, 3 = f32 + add1
template<int EPI>
__global__ __launch_bounds__(256) void k_gemm_wmma(
    const _Float16* __restrict__ A,   // [M,K] f16
    const _Float16* __restrict__ Bt,  // [N,K] f16 (B transposed)
    void* __restrict__ Cout,          // [M,N] f32 or f16
    int N, int K,
    const float* __restrict__ bias,   // [N]    (EPI==1)
    const float* __restrict__ add0,   // [M,N]  (EPI==1)
    const float* __restrict__ add1)   // [M,N]  (EPI==1 or 3)
{
    __shared__ alignas(16) _Float16 As[2][128][32];
    __shared__ alignas(16) _Float16 Bs[2][128][32];

    const int tid  = threadIdx.x;
    const int lane = tid & 31;
    const int wave = tid >> 5;
    const int wm   = wave & 3;     // 32-row sub-range
    const int wn   = wave >> 2;    // 64-col sub-range
    const int bm   = blockIdx.y * 128;
    const int bn   = blockIdx.x * 128;
    const int lh   = lane & 15;
    const bool hiL = lane >= 16;

    v8f acc[2][4];
    #pragma unroll
    for (int i = 0; i < 2; ++i)
        #pragma unroll
        for (int j = 0; j < 4; ++j)
            #pragma unroll
            for (int r = 0; r < 8; ++r) acc[i][j][r] = 0.0f;

    const int lr = tid >> 1;            // 0..127
    const int lc = (tid & 1) * 16;      // 0 or 16
    const _Float16* Abase = A  + (size_t)(bm + lr) * K + lc;
    const _Float16* Bbase = Bt + (size_t)(bn + lr) * K + lc;

    // prologue: fill buffer 0
    {
        async_ld_b128((unsigned)(uintptr_t)&As[0][lr][lc],     Abase);
        async_ld_b128((unsigned)(uintptr_t)&As[0][lr][lc + 8], Abase + 8);
        async_ld_b128((unsigned)(uintptr_t)&Bs[0][lr][lc],     Bbase);
        async_ld_b128((unsigned)(uintptr_t)&Bs[0][lr][lc + 8], Bbase + 8);
    }

    int buf = 0;
    for (int k0 = 0; k0 < K; k0 += 32) {
        wait_async0();          // this wave's fill of `buf` (and older) complete
        __syncthreads();        // all waves filled `buf`; all done reading buf^1
        if (k0 + 32 < K) {
            const int nb = buf ^ 1;
            async_ld_b128((unsigned)(uintptr_t)&As[nb][lr][lc],     Abase + k0 + 32);
            async_ld_b128((unsigned)(uintptr_t)&As[nb][lr][lc + 8], Abase + k0 + 40);
            async_ld_b128((unsigned)(uintptr_t)&Bs[nb][lr][lc],     Bbase + k0 + 32);
            async_ld_b128((unsigned)(uintptr_t)&Bs[nb][lr][lc + 8], Bbase + k0 + 40);
        }

        v16h afr[2], bfr[4];
        // A fragment: lanes 0-15 hold K{0..7,16..23}, lanes 16-31 K{8..15,24..31}
        #pragma unroll
        for (int i = 0; i < 2; ++i) {
            int r  = wm * 32 + i * 16 + lh;
            int kb = hiL ? 8 : 0;
            v8h x0 = *(const v8h*)&As[buf][r][kb];
            v8h x1 = *(const v8h*)&As[buf][r][kb + 16];
            #pragma unroll
            for (int j = 0; j < 8; ++j) { afr[i][j] = x0[j]; afr[i][j + 8] = x1[j]; }
        }
        // B fragment: lanes 0-15 hold K{0..7,16..23}, lanes 16-31 K{8..15,24..31}
        #pragma unroll
        for (int j = 0; j < 4; ++j) {
            int n  = wn * 64 + j * 16 + lh;
            int kb = hiL ? 8 : 0;
            v8h y0 = *(const v8h*)&Bs[buf][n][kb];
            v8h y1 = *(const v8h*)&Bs[buf][n][kb + 16];
            #pragma unroll
            for (int q = 0; q < 8; ++q) { bfr[j][q] = y0[q]; bfr[j][q + 8] = y1[q]; }
        }
        #pragma unroll
        for (int i = 0; i < 2; ++i)
            #pragma unroll
            for (int j = 0; j < 4; ++j)
                acc[i][j] = __builtin_amdgcn_wmma_f32_16x16x32_f16(
                    false, afr[i], false, bfr[j], (short)0, acc[i][j], false, false);
        buf ^= 1;
    }

    // C/D layout: VGPR r -> M = r (+8 for hi lanes), N = lane&15
    #pragma unroll
    for (int i = 0; i < 2; ++i) {
        int rowBase = bm + wm * 32 + i * 16 + (hiL ? 8 : 0);
        #pragma unroll
        for (int j = 0; j < 4; ++j) {
            int col = bn + wn * 64 + j * 16 + lh;
            #pragma unroll
            for (int r = 0; r < 8; ++r) {
                size_t idx = (size_t)(rowBase + r) * N + col;
                float v = acc[i][j][r];
                if (EPI == 1) v += bias[col] + add0[idx] + add1[idx];
                if (EPI == 3) v += add1[idx];
                if (EPI == 2) ((_Float16*)Cout)[idx] = (_Float16)gelu_exact(v);
                else          ((float*)Cout)[idx]    = v;
            }
        }
    }
}

// ---------------- attention stats: norms + Gram + softmax per (b,h) ----------------
// Raw-reshape semantics: q[b,h,d,n] = q_inp[b, h*2048 + n>>3, ((n&7)<<5) + d]
__global__ __launch_bounds__(256) void k_attn_stats(const float* __restrict__ qkv,
                                                    const float* __restrict__ rescale,
                                                    float* __restrict__ attn)
{
    __shared__ float kt[32][33];
    __shared__ float qt[32][33];
    __shared__ float Sm[32][33];
    __shared__ float kk[32];
    __shared__ float qq[32];

    const int bh = blockIdx.x;            // b*8 + h
    const int h  = bh & 7;
    const size_t rowBase = (size_t)(bh >> 3) * HW + h * 2048;

    const int t  = threadIdx.x;
    const int d  = t >> 3;
    const int e4 = (t & 7) * 4;

    float a0=0,a1=0,a2=0,a3=0, akk=0, aq0=0,aq1=0,aq2=0,aq3=0;

    for (int n0 = 0; n0 < HW; n0 += 32) {
        #pragma unroll
        for (int i = 0; i < 4; ++i) {
            int idx = t + i * 256;
            int nl = idx >> 5, dd = idx & 31;
            int n  = n0 + nl;
            size_t row = rowBase + (n >> 3);
            int col = ((n & 7) << 5) + dd;
            qt[nl][dd] = qkv[row * 768 + col];
            kt[nl][dd] = qkv[row * 768 + 256 + col];
        }
        __syncthreads();
        #pragma unroll
        for (int nn = 0; nn < 32; ++nn) {
            float kd = kt[nn][d];
            float q0 = qt[nn][e4 + 0];
            float q1 = qt[nn][e4 + 1];
            float q2 = qt[nn][e4 + 2];
            float q3 = qt[nn][e4 + 3];
            a0 += kd * q0; a1 += kd * q1; a2 += kd * q2; a3 += kd * q3;
            if ((t & 7) == 0) akk += kd * kd;
            if (d == 0) { aq0 += q0*q0; aq1 += q1*q1; aq2 += q2*q2; aq3 += q3*q3; }
        }
        __syncthreads();
    }
    Sm[d][e4+0]=a0; Sm[d][e4+1]=a1; Sm[d][e4+2]=a2; Sm[d][e4+3]=a3;
    if ((t & 7) == 0) kk[d] = akk;
    if (d == 0) { qq[e4+0]=aq0; qq[e4+1]=aq1; qq[e4+2]=aq2; qq[e4+3]=aq3; }
    __syncthreads();

    if (t < 32) {
        const float rs = rescale[h];
        const float kn = fmaxf(sqrtf(kk[t]), 1e-12f);
        float rowv[32];
        float mx = -3.4e38f;
        #pragma unroll
        for (int e = 0; e < 32; ++e) {
            float qn = fmaxf(sqrtf(qq[e]), 1e-12f);
            float a  = rs * Sm[t][e] / (kn * qn);
            rowv[e] = a;
            mx = fmaxf(mx, a);
        }
        float s = 0.f;
        #pragma unroll
        for (int e = 0; e < 32; ++e) { rowv[e] = expf(rowv[e] - mx); s += rowv[e]; }
        float inv = 1.f / s;
        #pragma unroll
        for (int e = 0; e < 32; ++e)
            attn[((size_t)bh * 32 + t) * 32 + e] = rowv[e] * inv;
    }
}

// ---------------- attention apply: o[b,p,cc] = sum_e attn[b,h,d,e] * (v*illu) ----------------
__global__ __launch_bounds__(256) void k_attn_apply(const float* __restrict__ qkv,
                                                    const float* __restrict__ illu,
                                                    const float* __restrict__ attn,
                                                    _Float16* __restrict__ outA)
{
    __shared__ float vls[256];
    const int t   = threadIdx.x;        // = output channel cc
    const int h   = t >> 5;
    const int pg0 = blockIdx.x * 64;
    const int b   = pg0 >> 14;
    float arow[32];
    {
        const float* ap = attn + ((size_t)(b * 8 + h) * 32 + (t & 31)) * 32;
        #pragma unroll
        for (int e = 0; e < 32; ++e) arow[e] = ap[e];
    }
    for (int pl = 0; pl < 64; ++pl) {
        const int pg = pg0 + pl;
        const int p  = pg & (HW - 1);
        const size_t row = (size_t)b * HW + h * 2048 + (p >> 3);
        const int col = ((p & 7) << 5) + (t & 31);
        vls[t] = qkv[row * 768 + 512 + col] * illu[row * 256 + col];
        __syncthreads();
        float o = 0.f;
        #pragma unroll
        for (int e = 0; e < 32; ++e) o += arow[e] * vls[(h << 5) + e];
        outA[(size_t)pg * 256 + t] = (_Float16)o;
        __syncthreads();
    }
}

// ---------------- depthwise 3x3, NHWC, SAME padding ----------------
template<int C, typename TIN, typename TOUT, bool GIN, bool GOUT>
__global__ __launch_bounds__(256) void k_dw3x3(const TIN* __restrict__ in, int ldin,
                                               const float* __restrict__ ker,
                                               TOUT* __restrict__ out, int ldout)
{
    constexpr int CS = (C == 256) ? 8 : 10;
    int gid = blockIdx.x * 256 + threadIdx.x;   // (b*HW + y*128 + x)*C + c
    int c   = gid & (C - 1);
    int pix = gid >> CS;                        // b*HW + y*128 + x
    int b   = pix >> 14;
    int yx  = pix & (HW - 1);
    int y   = yx >> 7, x = yx & 127;

    float acc = 0.f;
    #pragma unroll
    for (int ky = 0; ky < 3; ++ky) {
        int yy = y + ky - 1;
        if (yy < 0 || yy > 127) continue;
        #pragma unroll
        for (int kx = 0; kx < 3; ++kx) {
            int xx = x + kx - 1;
            if (xx < 0 || xx > 127) continue;
            float v = (float)in[((size_t)(b * HW + yy * 128 + xx)) * ldin + c];
            if (GIN) v = gelu_exact(v);
            acc += v * ker[(ky * 3 + kx) * C + c];
        }
    }
    if (GOUT) acc = gelu_exact(acc);
    out[(size_t)pix * ldout + c] = (TOUT)acc;
}

// ---------------- LayerNorm over C=256, f16 output ----------------
__global__ __launch_bounds__(256) void k_layernorm(const float* __restrict__ x,
                                                   const float* __restrict__ g,
                                                   const float* __restrict__ b2,
                                                   _Float16* __restrict__ out)
{
    __shared__ float s1[256];
    __shared__ float s2[256];
    const int t = threadIdx.x;
    const size_t row = blockIdx.x;
    float v = x[row * 256 + t];
    s1[t] = v; s2[t] = v * v;
    __syncthreads();
    for (int off = 128; off > 0; off >>= 1) {
        if (t < off) { s1[t] += s1[t + off]; s2[t] += s2[t + off]; }
        __syncthreads();
    }
    float mu  = s1[0] * (1.f / 256.f);
    float var = s2[0] * (1.f / 256.f) - mu * mu;
    float inv = rsqrtf(var + 1e-5f);
    out[row * 256 + t] = (_Float16)((v - mu) * inv * g[t] + b2[t]);
}

// ---------------- host orchestration ----------------
extern "C" void kernel_launch(void* const* d_in, const int* in_sizes, int n_in,
                              void* d_out, int out_size, void* d_ws, size_t ws_size,
                              hipStream_t stream)
{
    (void)in_sizes; (void)n_in; (void)out_size; (void)ws_size;
    const float* x    = (const float*)d_in[0];
    const float* illu = (const float*)d_in[1];
    const float* Wq   = (const float*)d_in[2];
    const float* Wk   = (const float*)d_in[3];
    const float* Wv   = (const float*)d_in[4];
    const float* resc = (const float*)d_in[5];
    const float* Wp   = (const float*)d_in[6];
    const float* bp   = (const float*)d_in[7];
    const float* pe1  = (const float*)d_in[8];
    const float* pe2  = (const float*)d_in[9];
    const float* lng  = (const float*)d_in[10];
    const float* lnb  = (const float*)d_in[11];
    const float* f1   = (const float*)d_in[12];
    const float* fdw  = (const float*)d_in[13];
    const float* f2   = (const float*)d_in[14];

    char* ws = (char*)d_ws;
    float*    xh    = (float*)(ws);                       // 33,554,432 B
    float*    illuh = (float*)(ws + 33554432);            // 33,554,432 B
    float*    qkv   = (float*)(ws + 67108864);            // 100,663,296 B  [M2,768]
    _Float16* AhA   = (_Float16*)(ws + 167772160);        // 67,108,864 B  [M2,1024] f16
    char*     AhBb  = ws + 234881024;                     // 67,108,864 B
    _Float16* AhB   = (_Float16*)AhBb;                    //   FFN: [M2,1024] f16
    float*    outp  = (float*)AhBb;                       //   attn phase: [M2,256] f32
    float*    tmp1  = (float*)(AhBb + 33554432);          //   attn phase: [M2,256] f32
    float*    attn  = (float*)(ws + 301989888);           // 65,536 B
    _Float16* Bt    = (_Float16*)(ws + 302055424);        // 524,288 B

    dim3 blk(256);

    k_nchw_to_nhwc<<<32768, blk, 0, stream>>>(x, xh);
    k_nchw_to_nhwc<<<32768, blk, 0, stream>>>(illu, illuh);

    for (int i = 0; i < NBLK; ++i) {
        // --- qkv GEMM (Wq|Wk|Wv fused, N=768) ---
        k_stage_wt<<<256, blk, 0, stream>>>(Wq + i * 65536, Bt,          256, 256);
        k_stage_wt<<<256, blk, 0, stream>>>(Wk + i * 65536, Bt + 65536,  256, 256);
        k_stage_wt<<<256, blk, 0, stream>>>(Wv + i * 65536, Bt + 131072, 256, 256);
        k_f32_to_f16<<<32768, blk, 0, stream>>>(xh, AhA, M2 * 256);
        k_gemm_wmma<0><<<dim3(6, 256), blk, 0, stream>>>(
            AhA, Bt, qkv, 768, 256, nullptr, nullptr, nullptr);

        // --- positional path on v_inp: dw3x3 -> gelu -> dw3x3 ---
        k_dw3x3<256, float, float, false, false><<<32768, blk, 0, stream>>>(
            qkv + 512, 768, pe1 + i * 2304, tmp1, 256);
        k_dw3x3<256, float, float, true, false><<<32768, blk, 0, stream>>>(
            tmp1, 256, pe2 + i * 2304, outp, 256);

        // --- attention (raw-reshape semantics) ---
        k_attn_stats<<<16, blk, 0, stream>>>(qkv, resc + i * 8, attn);
        k_attn_apply<<<512, blk, 0, stream>>>(qkv, illuh, attn, AhA);

        // --- proj GEMM + bp + out_p + residual -> xh ---
        k_stage_wt<<<256, blk, 0, stream>>>(Wp + i * 65536, Bt, 256, 256);
        k_gemm_wmma<1><<<dim3(2, 256), blk, 0, stream>>>(
            AhA, Bt, xh, 256, 256, bp + i * 256, outp, xh);

        // --- FFN: LN -> f1(+gelu) -> dw3x3(+gelu) -> f2 + residual ---
        k_layernorm<<<32768, blk, 0, stream>>>(xh, lng + i * 256, lnb + i * 256, AhA);
        k_stage_wt<<<1024, blk, 0, stream>>>(f1 + i * 262144, Bt, 256, 1024);
        k_gemm_wmma<2><<<dim3(8, 256), blk, 0, stream>>>(
            AhA, Bt, AhB, 1024, 256, nullptr, nullptr, nullptr);
        k_dw3x3<1024, _Float16, _Float16, false, true><<<131072, blk, 0, stream>>>(
            AhB, 1024, fdw + i * 9216, AhA, 1024);
        k_stage_wt<<<1024, blk, 0, stream>>>(f2 + i * 262144, Bt, 1024, 256);
        k_gemm_wmma<3><<<dim3(2, 256), blk, 0, stream>>>(
            AhA, Bt, xh, 256, 1024, nullptr, nullptr, xh);
    }

    k_nhwc_to_nchw<<<32768, blk, 0, stream>>>(xh, (float*)d_out);
}
